// Decoder_81277961109824
// MI455X (gfx1250) — matrix operand order repaired and checked by reading, not compiled, not run on hardware
//
#include <hip/hip_runtime.h>
#include <hip/hip_bf16.h>
#include <math.h>

// Model dims (fixed by the reference)
#define BB   2
#define SS   1024
#define DD   1024
#define HH   16
#define LLAY 4
#define VV   32000
#define DKK  64
#define BSR  (BB * SS)          // 2048 rows

typedef __attribute__((ext_vector_type(16))) __bf16         v16bf;
typedef __attribute__((ext_vector_type(8)))  float          v8f;
typedef __attribute__((ext_vector_type(16))) unsigned short v16u;
typedef __attribute__((ext_vector_type(8)))  unsigned short v8u;
typedef __attribute__((ext_vector_type(4)))  int            v4i;

#define AS1 __attribute__((address_space(1)))
#define AS3 __attribute__((address_space(3)))

#if __has_builtin(__builtin_amdgcn_global_load_async_to_lds_b128)
#define HAVE_ASYNC_LDS 1
#endif

// ---------------------------------------------------------------- helpers
__device__ __forceinline__ unsigned short f2bf(float f) {
  unsigned u = __float_as_uint(f);
  unsigned r = u + 0x7FFFu + ((u >> 16) & 1u);   // round-to-nearest-even
  return (unsigned short)(r >> 16);
}

__device__ __forceinline__ v16u cat8(v8u lo, v8u hi) {
  v16u r;
#pragma unroll
  for (int i = 0; i < 8; ++i) { r[i] = lo[i]; r[8 + i] = hi[i]; }
  return r;
}

__device__ __forceinline__ v8f wmma_bf16(v16u a, v16u b, v8f c) {
  return __builtin_amdgcn_wmma_f32_16x16x32_bf16(
      false, __builtin_bit_cast(v16bf, a),
      false, __builtin_bit_cast(v16bf, b),
      (short)0, c, false, false);
}

// 16-byte global -> LDS copy.  Uses CDNA5 async DMA (ASYNCcnt) when available;
// falls back to a VGPR round-trip copy otherwise.
__device__ __forceinline__ void async_cp16(const unsigned short* g,
                                           unsigned short* l) {
#ifdef HAVE_ASYNC_LDS
  __builtin_amdgcn_global_load_async_to_lds_b128(
      (AS1 v4i*)(void*)g, (AS3 v4i*)(void*)l, 0, 0);
#else
  *(v8u*)l = *(const v8u*)g;
#endif
}

__device__ __forceinline__ void wait_async0() {
#ifdef HAVE_ASYNC_LDS
#if __has_builtin(__builtin_amdgcn_s_wait_asynccnt)
  __builtin_amdgcn_s_wait_asynccnt(0);
#else
  asm volatile("s_wait_asynccnt 0" ::: "memory");
#endif
#endif
}

// ---------------------------------------------------------------- cvt f32 -> bf16
__global__ void cvt_bf16_kernel(const float* __restrict__ s,
                                unsigned short* __restrict__ d, int n) {
  for (int i = blockIdx.x * blockDim.x + threadIdx.x; i < n;
       i += gridDim.x * blockDim.x)
    d[i] = f2bf(s[i]);
}

// ---------------------------------------------------------------- embedding
__global__ void embed_kernel(const int* __restrict__ tokens,
                             const float* __restrict__ emb,
                             const float* __restrict__ pos,
                             float* __restrict__ xf,
                             unsigned short* __restrict__ xb) {
  const int n = BB * SS * DD;
  for (int i = blockIdx.x * blockDim.x + threadIdx.x; i < n;
       i += gridDim.x * blockDim.x) {
    int d = i % DD;
    int bs = i / DD;
    int s = bs % SS;
    int tok = tokens[bs];
    float v = emb[(size_t)tok * DD + d] + pos[(size_t)s * DD + d];
    xf[i] = v;
    xb[i] = f2bf(v);
  }
}

// ---------------------------------------------------------------- WMMA GEMM
// C[M,N] = A[M,K](bf16) @ B[K,N](bf16) + bias[N]   (f32 out)
// Block tile 128x128, K-step 32, 256 threads = 8 waves (2x4), double-buffered LDS.
// A tile staged by async DMA; B tile staged transposed (batched load + scatter).
#define GTM 128
#define GTN 128
#define GTK 32
#define LSTR 40   // 32 + 8 pad (keeps 16B alignment, spreads banks)

__global__ __launch_bounds__(256) void gemm_bias_bf16(
    const unsigned short* __restrict__ A,
    const unsigned short* __restrict__ Bw,
    const float* __restrict__ bias,
    float* __restrict__ C, int M, int N, int K) {
  __shared__ __align__(16) unsigned short As[2][GTM * LSTR];
  __shared__ __align__(16) unsigned short Bs[2][GTN * LSTR];  // transposed [n][k]

  const int tid = threadIdx.x;
  const int m0 = blockIdx.y * GTM;
  const int n0 = blockIdx.x * GTN;
  const int w = tid >> 5, lane = tid & 31;
  const int wr = w >> 2, wc = w & 3;       // 2 x 4 wave grid
  const int lh = lane >> 4, ll = lane & 15;

  v8f acc[4][2];
#pragma unroll
  for (int mt = 0; mt < 4; ++mt)
#pragma unroll
    for (int nt = 0; nt < 2; ++nt) acc[mt][nt] = (v8f){0.f, 0.f, 0.f, 0.f, 0.f, 0.f, 0.f, 0.f};

  // A tile: 512 16-byte chunks, DMA'd straight into LDS
  auto loadA = [&](int buf, int k0) {
#pragma unroll
    for (int i = 0; i < 2; ++i) {
      int c = tid + 256 * i;
      int row = c >> 2, seg = (c & 3) * 8;
      async_cp16(A + (size_t)(m0 + row) * K + k0 + seg,
                 &As[buf][row * LSTR + seg]);
    }
  };
  // B tile: batched global loads, then transpose-scatter into LDS
  auto loadB = [&](int buf, int k0) {
    v8u d[2];
    int kk[2], nn[2];
#pragma unroll
    for (int i = 0; i < 2; ++i) {
      int c = tid + 256 * i;
      kk[i] = c >> 4;                  // 0..31
      nn[i] = (c & 15) * 8;            // 0..120
      d[i] = *(const v8u*)(Bw + (size_t)(k0 + kk[i]) * N + n0 + nn[i]);
    }
#pragma unroll
    for (int i = 0; i < 2; ++i)
#pragma unroll
      for (int j = 0; j < 8; ++j) Bs[buf][(nn[i] + j) * LSTR + kk[i]] = d[i][j];
  };

  loadA(0, 0);
  loadB(0, 0);
  wait_async0();
  __syncthreads();

  const int nk = K / GTK;
  for (int ks = 0; ks < nk; ++ks) {
    int buf = ks & 1;
    if (ks + 1 < nk) {
      loadA(buf ^ 1, (ks + 1) * GTK);
      loadB(buf ^ 1, (ks + 1) * GTK);
      if (ks + 2 < nk)
        __builtin_prefetch(Bw + (size_t)((ks + 2) * GTK + (tid >> 4)) * N + n0, 0, 1);
    }

    v16u af[4], bfrag[2];
#pragma unroll
    for (int mt = 0; mt < 4; ++mt) {
      int row = wr * 64 + mt * 16 + ll;
      const unsigned short* base = &As[buf][row * LSTR];
      v8u lo = *(const v8u*)(base + lh * 8);
      v8u hi = *(const v8u*)(base + 16 + lh * 8);
      af[mt] = cat8(lo, hi);
    }
#pragma unroll
    for (int nt = 0; nt < 2; ++nt) {
      int n = wc * 32 + nt * 16 + ll;
      const unsigned short* base = &Bs[buf][n * LSTR];
      v8u lo = *(const v8u*)(base + lh * 8);
      v8u hi = *(const v8u*)(base + 16 + lh * 8);
      bfrag[nt] = cat8(lo, hi);
    }
#pragma unroll
    for (int mt = 0; mt < 4; ++mt)
#pragma unroll
      for (int nt = 0; nt < 2; ++nt)
        acc[mt][nt] = wmma_bf16(af[mt], bfrag[nt], acc[mt][nt]);

    wait_async0();           // next tile's DMA must land before the barrier
    __syncthreads();
  }

#pragma unroll
  for (int mt = 0; mt < 4; ++mt)
#pragma unroll
    for (int nt = 0; nt < 2; ++nt) {
      int n = n0 + wc * 32 + nt * 16 + ll;
      float bv = bias[n];
#pragma unroll
      for (int r = 0; r < 8; ++r) {
        int m = m0 + wr * 64 + mt * 16 + r + 8 * lh;
        C[(size_t)m * N + n] = acc[mt][nt][r] + bv;
      }
    }
}

// ---------------------------------------------------------------- QKV head split
// hf [B,S,3D] f32 -> q,k,v bf16 [B,H,S,DK]; q pre-scaled by 1/sqrt(DK)
__global__ void qkv_prep_kernel(const float* __restrict__ hf,
                                unsigned short* __restrict__ qb,
                                unsigned short* __restrict__ kb,
                                unsigned short* __restrict__ vb) {
  const int n = BB * SS * DD;
  for (int i = blockIdx.x * blockDim.x + threadIdx.x; i < n;
       i += gridDim.x * blockDim.x) {
    int d = i % DD;
    int bs = i / DD;
    int b = bs / SS, s = bs % SS;
    int h = d / DKK, dk = d % DKK;
    size_t src = (size_t)bs * (3 * DD);
    size_t dst = (((size_t)(b * HH + h)) * SS + s) * DKK + dk;
    qb[dst] = f2bf(hf[src + d] * 0.125f);  // 1/sqrt(64)
    kb[dst] = f2bf(hf[src + DD + d]);
    vb[dst] = f2bf(hf[src + 2 * DD + d]);
  }
}

// ---------------------------------------------------------------- flash attention
// grid (S/64, B*H), 128 threads = 4 waves; wave w owns 16 q-rows.
#define FSTR 72   // 64 + 8 pad
__global__ __launch_bounds__(128) void flash_attn_kernel(
    const unsigned short* __restrict__ q, const unsigned short* __restrict__ k,
    const unsigned short* __restrict__ v, const unsigned char* __restrict__ amask,
    unsigned short* __restrict__ ob) {
  __shared__ __align__(16) unsigned short Ks[64 * FSTR];       // [key][dk]
  __shared__ __align__(16) unsigned short Vt[64 * FSTR];       // [dk][key]
  __shared__ __align__(16) unsigned short Ps[4][16 * FSTR];    // per-wave P
  __shared__ float smask[64];

  const int tid = threadIdx.x;
  const int qt = blockIdx.x;            // q tile (64 rows)
  const int bh = blockIdx.y;
  const int b = bh / HH, h = bh % HH;
  const int w = tid >> 5, lane = tid & 31;
  const int lh = lane >> 4, ll = lane & 15;
  const size_t base = (size_t)bh * SS * DKK;

  // Q fragments (A-operand), loaded straight from global
  v16u aq[2];
  {
    int qrow = qt * 64 + w * 16 + ll;
#pragma unroll
    for (int kk = 0; kk < 2; ++kk) {
      const unsigned short* p = q + base + (size_t)qrow * DKK + kk * 32;
      v8u lo = *(const v8u*)(p + lh * 8);
      v8u hi = *(const v8u*)(p + 16 + lh * 8);
      aq[kk] = cat8(lo, hi);
    }
  }

  float mrow[8], lsum[8];
#pragma unroll
  for (int r = 0; r < 8; ++r) { mrow[r] = -1e30f; lsum[r] = 0.f; }
  v8f oacc[4];
#pragma unroll
  for (int nt = 0; nt < 4; ++nt) oacc[nt] = (v8f){0.f, 0.f, 0.f, 0.f, 0.f, 0.f, 0.f, 0.f};

  for (int j = 0; j <= qt; ++j) {
    // ---- stage K (async DMA, row-major) and V (batched load + transpose) ----
    {
      v8u dv[4];
      int row[4], seg[4];
#pragma unroll
      for (int i = 0; i < 4; ++i) {
        int c = tid + 128 * i;          // 512 chunks of 8
        row[i] = c >> 3;
        seg[i] = (c & 7) * 8;
        async_cp16(k + base + (size_t)(j * 64 + row[i]) * DKK + seg[i],
                   &Ks[row[i] * FSTR + seg[i]]);
        dv[i] = *(const v8u*)(v + base + (size_t)(j * 64 + row[i]) * DKK + seg[i]);
      }
#pragma unroll
      for (int i = 0; i < 4; ++i)
#pragma unroll
        for (int t2 = 0; t2 < 8; ++t2) Vt[(seg[i] + t2) * FSTR + row[i]] = dv[i][t2];
    }
    if (tid < 64)
      smask[tid] = amask[(size_t)b * SS + j * 64 + tid] ? 0.f : -1e30f;
    wait_async0();
    __syncthreads();

    // ---- S = Q K^T ----
    v8f sacc[4];
#pragma unroll
    for (int nt = 0; nt < 4; ++nt) sacc[nt] = (v8f){0.f, 0.f, 0.f, 0.f, 0.f, 0.f, 0.f, 0.f};
#pragma unroll
    for (int nt = 0; nt < 4; ++nt) {
      int key = nt * 16 + ll;
#pragma unroll
      for (int kk = 0; kk < 2; ++kk) {
        const unsigned short* bp = &Ks[key * FSTR + kk * 32];
        v8u lo = *(const v8u*)(bp + lh * 8);
        v8u hi = *(const v8u*)(bp + 16 + lh * 8);
        sacc[nt] = wmma_bf16(aq[kk], cat8(lo, hi), sacc[nt]);
      }
    }

    // ---- mask + online softmax ----
    float alpha[8];
#pragma unroll
    for (int r = 0; r < 8; ++r) {
      int qr = qt * 64 + w * 16 + r + 8 * lh;
      float mx = -1e30f;
#pragma unroll
      for (int nt = 0; nt < 4; ++nt) {
        float sv = sacc[nt][r] + smask[nt * 16 + ll];
        if (j == qt) {
          int key = j * 64 + nt * 16 + ll;
          if (key > qr) sv = -1e30f;
        }
        sacc[nt][r] = sv;
        mx = fmaxf(mx, sv);
      }
#pragma unroll
      for (int o = 1; o < 16; o <<= 1) mx = fmaxf(mx, __shfl_xor(mx, o, 16));
      float mnew = fmaxf(mrow[r], mx);
      float a = __expf(mrow[r] - mnew);
      alpha[r] = a;
      float rs = 0.f;
#pragma unroll
      for (int nt = 0; nt < 4; ++nt) {
        float p = __expf(sacc[nt][r] - mnew);
        sacc[nt][r] = p;
        rs += p;
      }
#pragma unroll
      for (int o = 1; o < 16; o <<= 1) rs += __shfl_xor(rs, o, 16);
      lsum[r] = lsum[r] * a + rs;
      mrow[r] = mnew;
    }
#pragma unroll
    for (int nt = 0; nt < 4; ++nt)
#pragma unroll
      for (int r = 0; r < 8; ++r) oacc[nt][r] *= alpha[r];

    // ---- P -> LDS (bf16), refragment as A-operand ----
#pragma unroll
    for (int nt = 0; nt < 4; ++nt)
#pragma unroll
      for (int r = 0; r < 8; ++r)
        Ps[w][(r + 8 * lh) * FSTR + nt * 16 + ll] = f2bf(sacc[nt][r]);
    __syncthreads();

    v16u ap[2];
#pragma unroll
    for (int kk = 0; kk < 2; ++kk) {
      const unsigned short* pp = &Ps[w][ll * FSTR + kk * 32];
      v8u lo = *(const v8u*)(pp + lh * 8);
      v8u hi = *(const v8u*)(pp + 16 + lh * 8);
      ap[kk] = cat8(lo, hi);
    }

    // ---- O += P V ----
#pragma unroll
    for (int nt = 0; nt < 4; ++nt) {
      int dk = nt * 16 + ll;
#pragma unroll
      for (int kk = 0; kk < 2; ++kk) {
        const unsigned short* bp = &Vt[dk * FSTR + kk * 32];
        v8u lo = *(const v8u*)(bp + lh * 8);
        v8u hi = *(const v8u*)(bp + 16 + lh * 8);
        oacc[nt] = wmma_bf16(ap[kk], cat8(lo, hi), oacc[nt]);
      }
    }
    __syncthreads();
  }

  // ---- normalize and write merged heads (bf16) ----
#pragma unroll
  for (int nt = 0; nt < 4; ++nt)
#pragma unroll
    for (int r = 0; r < 8; ++r) {
      int row = qt * 64 + w * 16 + r + 8 * lh;
      int dk = nt * 16 + ll;
      float val = oacc[nt][r] / lsum[r];
      ob[((size_t)b * SS + row) * DD + h * DKK + dk] = f2bf(val);
    }
}

// ---------------------------------------------------------------- residual + LN
__global__ __launch_bounds__(256) void resid_ln_kernel(
    const float* __restrict__ oproj, float* __restrict__ xf,
    unsigned short* __restrict__ xb, const float* __restrict__ g,
    const float* __restrict__ bln) {
  __shared__ float red[256];
  const int row = blockIdx.x, t = threadIdx.x;
  const float* op = oproj + (size_t)row * DD;
  float* xr = xf + (size_t)row * DD;
  float loc[4];
  float s = 0.f;
#pragma unroll
  for (int i = 0; i < 4; ++i) {
    int c = t + i * 256;
    float v = op[c] + xr[c];
    loc[i] = v;
    s += v;
  }
  red[t] = s;
  __syncthreads();
  for (int o = 128; o > 0; o >>= 1) { if (t < o) red[t] += red[t + o]; __syncthreads(); }
  float mean = red[0] / DD;
  __syncthreads();
  s = 0.f;
#pragma unroll
  for (int i = 0; i < 4; ++i) { float d = loc[i] - mean; s += d * d; }
  red[t] = s;
  __syncthreads();
  for (int o = 128; o > 0; o >>= 1) { if (t < o) red[t] += red[t + o]; __syncthreads(); }
  float inv = rsqrtf(red[0] / DD + 1e-5f);
#pragma unroll
  for (int i = 0; i < 4; ++i) {
    int c = t + i * 256;
    float y = (loc[i] - mean) * inv * g[c] + bln[c];
    xr[c] = y;
    xb[(size_t)row * DD + c] = f2bf(y);
  }
}

// ---------------------------------------------------------------- GEGLU + LN
__global__ __launch_bounds__(256) void geglu_ln_kernel(
    const float* __restrict__ h2, float* __restrict__ xf,
    unsigned short* __restrict__ xb, const float* __restrict__ g,
    const float* __restrict__ bln) {
  __shared__ float red[256];
  const int row = blockIdx.x, t = threadIdx.x;
  const float* hr = h2 + (size_t)row * (2 * DD);
  float loc[4];
  float s = 0.f;
#pragma unroll
  for (int i = 0; i < 4; ++i) {
    int c = t + i * 256;
    float a = hr[c];
    float gg = hr[DD + c];
    float gel = 0.5f * gg * (1.f + erff(gg * 0.70710678118f));
    float v = a * gel;
    loc[i] = v;
    s += v;
  }
  red[t] = s;
  __syncthreads();
  for (int o = 128; o > 0; o >>= 1) { if (t < o) red[t] += red[t + o]; __syncthreads(); }
  float mean = red[0] / DD;
  __syncthreads();
  s = 0.f;
#pragma unroll
  for (int i = 0; i < 4; ++i) { float d = loc[i] - mean; s += d * d; }
  red[t] = s;
  __syncthreads();
  for (int o = 128; o > 0; o >>= 1) { if (t < o) red[t] += red[t + o]; __syncthreads(); }
  float inv = rsqrtf(red[0] / DD + 1e-5f);
#pragma unroll
  for (int i = 0; i < 4; ++i) {
    int c = t + i * 256;
    float y = (loc[i] - mean) * inv * g[c] + bln[c];
    xf[(size_t)row * DD + c] = y;
    xb[(size_t)row * DD + c] = f2bf(y);
  }
}

// ---------------------------------------------------------------- launch
extern "C" void kernel_launch(void* const* d_in, const int* in_sizes, int n_in,
                              void* d_out, int out_size, void* d_ws, size_t ws_size,
                              hipStream_t stream) {
  const int* tokens = (const int*)d_in[0];
  const unsigned char* amask = (const unsigned char*)d_in[1];  // bool array
  const float* emb = (const float*)d_in[2];
  const float* pos = (const float*)d_in[3];
  const float* qkv_w = (const float*)d_in[4];
  const float* qkv_b = (const float*)d_in[5];
  const float* out_w = (const float*)d_in[6];
  const float* out_b = (const float*)d_in[7];
  const float* ln1_g = (const float*)d_in[8];
  const float* ln1_b = (const float*)d_in[9];
  const float* mlp_w = (const float*)d_in[10];
  const float* mlp_b = (const float*)d_in[11];
  const float* ln2_g = (const float*)d_in[12];
  const float* ln2_b = (const float*)d_in[13];
  const float* proj_w = (const float*)d_in[14];
  const float* proj_b = (const float*)d_in[15];
  float* out = (float*)d_out;

  char* ws = (char*)d_ws;
  size_t off = 0;
  auto alloc = [&](size_t bytes) -> char* {
    off = (off + 255) & ~(size_t)255;
    char* p = ws + off;
    off += bytes;
    return p;
  };

  const size_t nQKVW = (size_t)LLAY * DD * 3 * DD;
  const size_t nOUTW = (size_t)LLAY * DD * DD;
  const size_t nMLPW = (size_t)LLAY * DD * 2 * DD;
  const size_t nPRJW = (size_t)DD * VV;

  unsigned short* wqkv = (unsigned short*)alloc(nQKVW * 2);
  unsigned short* wout = (unsigned short*)alloc(nOUTW * 2);
  unsigned short* wmlp = (unsigned short*)alloc(nMLPW * 2);
  unsigned short* wprj = (unsigned short*)alloc(nPRJW * 2);
  float* xf = (float*)alloc((size_t)BSR * DD * 4);
  unsigned short* xb = (unsigned short*)alloc((size_t)BSR * DD * 2);
  float* hf = (float*)alloc((size_t)BSR * 3 * DD * 4);   // reused: qkv / oproj / mlp
  unsigned short* qb = (unsigned short*)alloc((size_t)BSR * DD * 2);
  unsigned short* kb = (unsigned short*)alloc((size_t)BSR * DD * 2);
  unsigned short* vb = (unsigned short*)alloc((size_t)BSR * DD * 2);
  unsigned short* ob = (unsigned short*)alloc((size_t)BSR * DD * 2);

  auto cvt = [&](const float* s, unsigned short* d, size_t n) {
    size_t blocks = (n + 255) / 256;
    if (blocks > 16384) blocks = 16384;
    cvt_bf16_kernel<<<dim3((unsigned)blocks), dim3(256), 0, stream>>>(s, d, (int)n);
  };
  cvt(qkv_w, wqkv, nQKVW);
  cvt(out_w, wout, nOUTW);
  cvt(mlp_w, wmlp, nMLPW);
  cvt(proj_w, wprj, nPRJW);

  embed_kernel<<<dim3(4096), dim3(256), 0, stream>>>(tokens, emb, pos, xf, xb);

  auto gemm = [&](const unsigned short* A, const unsigned short* B,
                  const float* bias, float* C, int M, int N, int K) {
    gemm_bias_bf16<<<dim3(N / GTN, M / GTM), dim3(256), 0, stream>>>(A, B, bias, C, M, N, K);
  };

  for (int l = 0; l < LLAY; ++l) {
    // QKV projection
    gemm(xb, wqkv + (size_t)l * DD * 3 * DD, qkv_b + (size_t)l * 3 * DD, hf,
         BSR, 3 * DD, DD);
    qkv_prep_kernel<<<dim3(4096), dim3(256), 0, stream>>>(hf, qb, kb, vb);
    // fused attention (WMMA QK^T + PV, online softmax)
    flash_attn_kernel<<<dim3(SS / 64, BB * HH), dim3(128), 0, stream>>>(qb, kb, vb, amask, ob);
    // output projection
    gemm(ob, wout + (size_t)l * DD * DD, out_b + (size_t)l * DD, hf, BSR, DD, DD);
    // post-LN residual
    resid_ln_kernel<<<dim3(BSR), dim3(256), 0, stream>>>(
        hf, xf, xb, ln1_g + (size_t)l * DD, ln1_b + (size_t)l * DD);
    // GEGLU MLP
    gemm(xb, wmlp + (size_t)l * DD * 2 * DD, mlp_b + (size_t)l * 2 * DD, hf,
         BSR, 2 * DD, DD);
    geglu_ln_kernel<<<dim3(BSR), dim3(256), 0, stream>>>(
        hf, xf, xb, ln2_g + (size_t)l * DD, ln2_b + (size_t)l * DD);
  }

  // final vocab projection (dominant GEMM: 2048 x 32000 x 1024)
  gemm(xb, wprj, proj_b, out, BSR, VV, DD);
}
